// SpeakerEmbeddingsModule_56221121904735
// MI455X (gfx1250) — matrix operand order repaired and checked by reading, not compiled, not run on hardware
//
#include <hip/hip_runtime.h>
#include <hip/hip_bf16.h>

// ---------------- problem dims ----------------
#define B_ROWS 65536
#define E_DIM  768
#define H_DIM  768
#define F_DIM  1536   // E + H
#define D_DIM  2048
#define N_TAB  262144

// ---------------- vector types ----------------
typedef __attribute__((ext_vector_type(16))) __bf16 v16bf;
typedef __attribute__((ext_vector_type(8)))  __bf16 v8bf;
typedef __attribute__((ext_vector_type(4)))  __bf16 v4bf;
typedef __attribute__((ext_vector_type(8)))  float  v8f;
typedef __attribute__((ext_vector_type(4)))  float  v4f;

// ---------------- tiling ----------------
#define TM 32                 // batch rows per workgroup
#define XS (F_DIM + 8)        // x-tile row stride in bf16 elems (skewed: 772 dw % 64 = 4)
#define HS (D_DIM + 8)        // h-tile row stride in bf16 elems (1028 dw % 64 = 4)
#define YS (H_DIM + 4)        // y-tile row stride in f32  elems (772 dw  % 64 = 4)
#define LDS_H_BYTES (TM * HS * 2)              // 131584
#define LDS_X_BYTES (TM * XS * 2)              // 98816 (>= TM*YS*4 = 98816)
#define LDS_BYTES   (LDS_H_BYTES + LDS_X_BYTES) // 230400 < 320KB/WGP

// ---------------- WMMA helpers ----------------
__device__ __forceinline__ v8f wmma_bf16(v16bf a, v16bf b, v8f c) {
    // D = A(16x32 bf16) * B(32x16 bf16) + C(16x16 f32)
    return __builtin_amdgcn_wmma_f32_16x16x32_bf16(
        /*neg_a=*/false, a, /*neg_b=*/false, b,
        /*c_mod=*/(short)0, c, /*reuse_a=*/false, /*reuse_b=*/false);
}

// A operand: 16x32 bf16 tile, row-major [M][K] in LDS.
// lane L holds row M = L&15; lane-lo: K = kk+0..7 and kk+16..23; lane-hi: +8 each.
__device__ __forceinline__ v16bf load_A16(const __bf16* base, int stride, int lane, int kk) {
    const __bf16* p = base + (lane & 15) * stride + kk + ((lane >> 4) << 3);
    v8bf a = *(const v8bf*)p;
    v8bf b = *(const v8bf*)(p + 16);
    v16bf r;
#pragma unroll
    for (int i = 0; i < 8; ++i) { r[i] = a[i]; r[i + 8] = b[i]; }
    return r;
}

// B operand: 32x16 bf16 tile; memory is [N][K] row-major (weights are [out][in]).
// lane L holds column N; lane-lo: K = kk+0..7 and kk+16..23; lane-hi: +8 each.
// caller passes p = base + rowN*stride + kk + ((lane>>4)<<3)
__device__ __forceinline__ v16bf load_B16(const __bf16* p) {
    v8bf a = *(const v8bf*)p;
    v8bf b = *(const v8bf*)(p + 16);
    v16bf r;
#pragma unroll
    for (int i = 0; i < 8; ++i) { r[i] = a[i]; r[i + 8] = b[i]; }
    return r;
}

__device__ __forceinline__ float gelu_exact(float z) {
    return 0.5f * z * (1.0f + erff(z * 0.70710678118654752f));
}

// ---------------- weight conversion f32 -> bf16 ----------------
__global__ void cvt_weights(const float* __restrict__ W1, const float* __restrict__ W2,
                            __bf16* __restrict__ W1b, __bf16* __restrict__ W2b) {
    const int n1 = D_DIM * F_DIM;   // 3145728 (mult of 4)
    const int n2 = H_DIM * D_DIM;   // 1572864 (mult of 4)
    const int total4 = (n1 + n2) >> 2;
    for (int j = blockIdx.x * blockDim.x + threadIdx.x; j < total4;
         j += gridDim.x * blockDim.x) {
        int e = j << 2;
        const float* src;
        __bf16* dst;
        if (e < n1) { src = W1 + e; dst = W1b + e; }
        else        { src = W2 + (e - n1); dst = W2b + (e - n1); }
        v4f f = *(const v4f*)src;
        v4bf o;
        o[0] = (__bf16)f[0]; o[1] = (__bf16)f[1];
        o[2] = (__bf16)f[2]; o[3] = (__bf16)f[3];
        *(v4bf*)dst = o;
    }
}

// ---------------- fused gather + MLP + LayerNorm + scatter ----------------
__global__ __launch_bounds__(256)
void speaker_mlp(const float* __restrict__ u_t,
                 const int*   __restrict__ s_i,
                 const float* __restrict__ emb,
                 const __bf16* __restrict__ W1b,
                 const float* __restrict__ b1,
                 const __bf16* __restrict__ W2b,
                 const float* __restrict__ b2,
                 const float* __restrict__ gam,
                 const float* __restrict__ bet,
                 float* __restrict__ out_h,    // [B, H]
                 float* __restrict__ upd)      // [N, H] (pre-copied table)
{
    extern __shared__ char smem[];
    __bf16* hs = (__bf16*)smem;                       // [TM][HS] bf16
    __bf16* xs = (__bf16*)(smem + LDS_H_BYTES);       // [TM][XS] bf16 (stages 0/1)
    float*  ys = (float*)(smem + LDS_H_BYTES);        // [TM][YS] f32  (stages 2/3)

    const int tid  = threadIdx.x;
    const int lane = tid & 31;
    const int wave = tid >> 5;
    const int b0   = blockIdx.x * TM;
    const int bgrp = (lane >> 4) << 3;   // B-operand K granule offset

    // ---- Stage 0: gather + concat -> x-tile (bf16) ----
    // 32 rows x 192 chunks of 8 floats
    for (int c = tid; c < TM * (F_DIM / 8); c += 256) {
        int r   = c / (F_DIM / 8);
        int col = (c % (F_DIM / 8)) * 8;
        int b   = b0 + r;
        const float* src = (col < E_DIM)
            ? (u_t + (size_t)b * E_DIM + col)
            : (emb + (size_t)s_i[b] * H_DIM + (col - E_DIM));
        v4f f0 = *(const v4f*)src;
        v4f f1 = *(const v4f*)(src + 4);
        v8bf o;
        o[0] = (__bf16)f0[0]; o[1] = (__bf16)f0[1];
        o[2] = (__bf16)f0[2]; o[3] = (__bf16)f0[3];
        o[4] = (__bf16)f1[0]; o[5] = (__bf16)f1[1];
        o[6] = (__bf16)f1[2]; o[7] = (__bf16)f1[3];
        *(v8bf*)(xs + r * XS + col) = o;
    }
    __syncthreads();

    // ---- Stage 1: h = GELU(x @ W1^T + b1) -> h-tile (bf16) ----
    // 128 n-tiles over D=2048; wave owns 16, processes 2 at a time.
    for (int j2 = 0; j2 < 8; ++j2) {
        const int nt0 = wave * 16 + j2 * 2;
        const int nt1 = nt0 + 1;
        const int d0  = nt0 * 16 + (lane & 15);
        const int d1  = nt1 * 16 + (lane & 15);
        const __bf16* w0 = W1b + (size_t)d0 * F_DIM + bgrp;
        const __bf16* w1 = W1b + (size_t)d1 * F_DIM + bgrp;
        v8f acc00 = {}, acc01 = {}, acc10 = {}, acc11 = {};
#pragma unroll 4
        for (int kk = 0; kk < F_DIM; kk += 32) {
            v16bf A0 = load_A16(xs, XS, lane, kk);
            v16bf A1 = load_A16(xs + 16 * XS, XS, lane, kk);
            v16bf B0 = load_B16(w0 + kk);
            v16bf B1 = load_B16(w1 + kk);
            acc00 = wmma_bf16(A0, B0, acc00);
            acc01 = wmma_bf16(A0, B1, acc01);
            acc10 = wmma_bf16(A1, B0, acc10);
            acc11 = wmma_bf16(A1, B1, acc11);
        }
        // epilogue: bias + GELU -> hs ; C/D layout: M = v + 8*(lane>>4), N = lane&15
        const int mbase = (lane >> 4) << 3;
        const int n0 = nt0 * 16 + (lane & 15);
        const int n1 = nt1 * 16 + (lane & 15);
        const float bb0 = b1[n0], bb1 = b1[n1];
#pragma unroll
        for (int v = 0; v < 8; ++v) {
            int m = mbase + v;
            hs[m * HS + n0]        = (__bf16)gelu_exact(acc00[v] + bb0);
            hs[m * HS + n1]        = (__bf16)gelu_exact(acc01[v] + bb1);
            hs[(m + 16) * HS + n0] = (__bf16)gelu_exact(acc10[v] + bb0);
            hs[(m + 16) * HS + n1] = (__bf16)gelu_exact(acc11[v] + bb1);
        }
    }
    __syncthreads();

    // ---- Stage 2: y = h @ W2^T + b2 -> y-tile (f32, overlays x) ----
    // 48 n-tiles over H=768; wave owns 6, processes 2 at a time.
    for (int j2 = 0; j2 < 3; ++j2) {
        const int nt0 = wave * 6 + j2 * 2;
        const int nt1 = nt0 + 1;
        const int d0  = nt0 * 16 + (lane & 15);
        const int d1  = nt1 * 16 + (lane & 15);
        const __bf16* w0 = W2b + (size_t)d0 * D_DIM + bgrp;
        const __bf16* w1 = W2b + (size_t)d1 * D_DIM + bgrp;
        v8f acc00 = {}, acc01 = {}, acc10 = {}, acc11 = {};
#pragma unroll 4
        for (int kk = 0; kk < D_DIM; kk += 32) {
            v16bf A0 = load_A16(hs, HS, lane, kk);
            v16bf A1 = load_A16(hs + 16 * HS, HS, lane, kk);
            v16bf B0 = load_B16(w0 + kk);
            v16bf B1 = load_B16(w1 + kk);
            acc00 = wmma_bf16(A0, B0, acc00);
            acc01 = wmma_bf16(A0, B1, acc01);
            acc10 = wmma_bf16(A1, B0, acc10);
            acc11 = wmma_bf16(A1, B1, acc11);
        }
        const int mbase = (lane >> 4) << 3;
        const int n0 = nt0 * 16 + (lane & 15);
        const int n1 = nt1 * 16 + (lane & 15);
        const float bb0 = b2[n0], bb1 = b2[n1];
#pragma unroll
        for (int v = 0; v < 8; ++v) {
            int m = mbase + v;
            ys[m * YS + n0]        = acc00[v] + bb0;
            ys[m * YS + n1]        = acc01[v] + bb1;
            ys[(m + 16) * YS + n0] = acc10[v] + bb0;
            ys[(m + 16) * YS + n1] = acc11[v] + bb1;
        }
    }
    __syncthreads();

    // ---- Stage 3: LayerNorm per row + write new_h_t + scatter-add ----
    for (int r = wave; r < TM; r += 8) {
        const float* yr = ys + r * YS;
        float s = 0.f, ss = 0.f;
#pragma unroll
        for (int i = 0; i < H_DIM / 32; ++i) {
            float v = yr[lane + 32 * i];
            s += v; ss += v * v;
        }
#pragma unroll
        for (int m = 16; m >= 1; m >>= 1) {
            s  += __shfl_xor(s,  m, 32);
            ss += __shfl_xor(ss, m, 32);
        }
        const float mu  = s * (1.0f / H_DIM);
        const float var = ss * (1.0f / H_DIM) - mu * mu;
        const float inv = rsqrtf(var + 1e-5f);
        const int b = b0 + r;
        const size_t orow = (size_t)b * H_DIM;
        const size_t urow = (size_t)s_i[b] * H_DIM;
#pragma unroll
        for (int i = 0; i < H_DIM / 32; ++i) {
            int c = lane + 32 * i;
            float o = (yr[c] - mu) * inv * gam[c] + bet[c];
            out_h[orow + c] = o;
            unsafeAtomicAdd(&upd[urow + c], o);  // duplicates accumulate (HW f32 atomic)
        }
    }
}

// ---------------- launch ----------------
extern "C" void kernel_launch(void* const* d_in, const int* in_sizes, int n_in,
                              void* d_out, int out_size, void* d_ws, size_t ws_size,
                              hipStream_t stream) {
    const float* u_t = (const float*)d_in[0];
    const int*   s_i = (const int*)d_in[1];
    const float* emb = (const float*)d_in[2];
    const float* W1  = (const float*)d_in[3];
    const float* b1  = (const float*)d_in[4];
    const float* W2  = (const float*)d_in[5];
    const float* b2  = (const float*)d_in[6];
    const float* gam = (const float*)d_in[7];
    const float* bet = (const float*)d_in[8];

    float* out_h = (float*)d_out;
    float* upd   = out_h + (size_t)B_ROWS * H_DIM;

    __bf16* W1b = (__bf16*)d_ws;
    __bf16* W2b = W1b + (size_t)D_DIM * F_DIM;

    // allow 225KB dynamic LDS
    hipFuncSetAttribute((const void*)speaker_mlp,
                        hipFuncAttributeMaxDynamicSharedMemorySize, LDS_BYTES);

    // 1) bf16 weight staging (L2-resident afterwards: 9.4 MB << 192 MB)
    cvt_weights<<<2304, 256, 0, stream>>>(W1, W2, W1b, W2b);
    // 2) base table copy for the scatter-add output
    hipMemcpyAsync(upd, emb, (size_t)N_TAB * H_DIM * sizeof(float),
                   hipMemcpyDeviceToDevice, stream);
    // 3) fused gather + MLP(WMMA bf16) + LayerNorm + scatter
    speaker_mlp<<<B_ROWS / TM, 256, LDS_BYTES, stream>>>(
        u_t, s_i, emb, W1b, b1, W2b, b2, gam, bet, out_h, upd);
}
